// CompetitiveActor_86835648791097
// MI455X (gfx1250) — matrix operand order repaired and checked by reading, not compiled, not run on hardware
//
#include <hip/hip_runtime.h>
#include <hip/hip_bf16.h>

typedef __attribute__((ext_vector_type(16))) __bf16 v16bf;
typedef __attribute__((ext_vector_type(8)))  float  v8f;

#define B_     256
#define S_     512
#define OBS_   64
#define ZD_    16
#define HID_   256
#define ITERS_ 4
#define ELITES_ 64
#define NWAVE  8
#define ROWT   4            // 16-row tiles per workgroup -> 64 rows/WG
#define TM     (ROWT*16)

// ---------------- workspace layout ----------------
// bf16 (ushort) weights, transposed [N][Kpad], K padded to a multiple of 32
#define OFF_ENCW1 0
#define OFF_ENCW2 (OFF_ENCW1 + 256*64)
#define OFF_ZMU   (OFF_ENCW2 + 256*256)
#define OFF_ZLS   (OFF_ZMU   + 16*256)
#define OFF_MUW1  (OFF_ZLS   + 16*256)
#define OFF_MUW2  (OFF_MUW1  + 256*96)
#define OFF_MUW3  (OFF_MUW2  + 256*256)     /* cols 112..127 of mu_w3 */
#define OFF_SIGW1 (OFF_MUW3  + 16*256)
#define OFF_SIGW2 (OFF_SIGW1 + 256*32)
#define OFF_SIGW3 (OFF_SIGW2 + 256*256)
#define OFF_RKW1  (OFF_SIGW3 + 128*256)
#define OFF_RKW2  (OFF_RKW1  + 256*96)
#define OFF_RKW3  (OFF_RKW2  + 256*256)     /* col 0 of risk_w3, rest zero */
#define USH_TOTAL (OFF_RKW3  + 16*256)
// f32 region (float element offsets, base = d_ws + USH_TOTAL*2 bytes)
#define FOFF_MU    0
#define FOFF_STD   (FOFF_MU    + B_*ZD_)
#define FOFF_BESTZ (FOFF_STD   + B_*ZD_)
#define FOFF_BESTS (FOFF_BESTZ + B_*ZD_)
#define FOFF_SCORE (FOFF_BESTS + B_)

// ---------------- helpers ----------------
__device__ __forceinline__ unsigned short f2bf(float f) {
    union { float f; unsigned int u; } c; c.f = f;
    unsigned int u = c.u;
    u += 0x7FFFu + ((u >> 16) & 1u);          // round-to-nearest-even
    return (unsigned short)(u >> 16);
}

__device__ __forceinline__ v8f wmma_bf(v16bf a, v16bf b, v8f c) {
    return __builtin_amdgcn_wmma_f32_16x16x32_bf16(
        false, a, false, b, (short)0, c, false, false);
}

// A fragment (16x32, bf16) from row-major LDS tile with row stride kpad halfs.
// lane&15 = row M, halves 0..7 = K[kb+kbase .. +7], halves 8..15 = K[kb+16+kbase .. +7]
__device__ __forceinline__ v16bf ld_a(const unsigned short* a, int kpad, int kb, int lane) {
    int m = lane & 15;
    int kbase = kb + ((lane >> 4) << 3);
    const unsigned short* p = a + m * kpad + kbase;
    union { uint4 q[2]; v16bf v; } u;
    u.q[0] = *(const uint4*)(p);
    u.q[1] = *(const uint4*)(p + 16);
    return u.v;
}

// B fragment (32x16, bf16) from transposed weights WT[N][kpad]:
// lane&15 = column n, halves 0..7 = K[kb+(lane>>4)*8 .. +7], halves 8..15 = K[kb+16+(lane>>4)*8 .. +7]
__device__ __forceinline__ v16bf ld_b(const unsigned short* wt, int kpad, int n0, int kb, int lane) {
    int n = n0 + (lane & 15);
    int kk = kb + ((lane >> 4) << 3);
    const unsigned short* p = wt + (size_t)n * kpad + kk;
    union { uint4 q[2]; v16bf v; } u;
    u.q[0] = *(const uint4*)(p);
    u.q[1] = *(const uint4*)(p + 16);
    return u.v;
}

// OUT[TM][N] = relu(A[TM][kpad] @ WT^T + bias), bf16 out to LDS (stride ldo)
__device__ void layer_relu(const unsigned short* aLds, const unsigned short* wt,
                           const float* bias, unsigned short* outLds,
                           int kpad, int kt, int N, int ldo, int lane, int wave) {
    for (int t = wave; t < (N >> 4); t += NWAVE) {
        int n0 = t << 4;
        v8f acc[ROWT] = {};
        for (int kb = 0; kb < kt; ++kb) {
            int k0 = kb << 5;
            v16bf bf = ld_b(wt, kpad, n0, k0, lane);
#pragma unroll
            for (int rt = 0; rt < ROWT; ++rt) {
                v16bf af = ld_a(aLds + rt * 16 * kpad, kpad, k0, lane);
                acc[rt] = wmma_bf(af, bf, acc[rt]);
            }
        }
        int n  = n0 + (lane & 15);
        int mb = (lane >> 4) << 3;
        float bv = bias[n];
#pragma unroll
        for (int rt = 0; rt < ROWT; ++rt)
#pragma unroll
            for (int r = 0; r < 8; ++r) {
                float v = acc[rt][r] + bv;
                v = fmaxf(v, 0.f);
                outLds[(size_t)(rt * 16 + mb + r) * ldo + n] = f2bf(v);
            }
    }
}

// One 16-col head tile (f32 out, +bias, no relu), one row-tile per calling wave
__device__ void head_f32(const unsigned short* aLds, const unsigned short* wt,
                         const float* bias, int biasN, float* outLds /*[TM][16]*/,
                         int kpad, int kt, int rt, int lane) {
    v8f acc = {};
    for (int kb = 0; kb < kt; ++kb) {
        int k0 = kb << 5;
        v16bf bf = ld_b(wt, kpad, 0, k0, lane);
        v16bf af = ld_a(aLds + rt * 16 * kpad, kpad, k0, lane);
        acc = wmma_bf(af, bf, acc);
    }
    int n  = lane & 15;
    int mb = (lane >> 4) << 3;
    float bv = (n < biasN) ? bias[n] : 0.f;
#pragma unroll
    for (int r = 0; r < 8; ++r)
        outLds[(rt * 16 + mb + r) * 16 + n] = acc[r] + bv;
}

// sig3: N=128 (one 16-col tile per wave), clip to [-4,3], row-sum into logsum
__device__ void layer_sig3(const unsigned short* aLds, const unsigned short* wt,
                           const float* bias, float* logsum, int lane, int wave) {
    int n0 = wave << 4;
    v8f acc[ROWT] = {};
    for (int kb = 0; kb < 8; ++kb) {
        int k0 = kb << 5;
        v16bf bf = ld_b(wt, 256, n0, k0, lane);
#pragma unroll
        for (int rt = 0; rt < ROWT; ++rt) {
            v16bf af = ld_a(aLds + rt * 16 * 256, 256, k0, lane);
            acc[rt] = wmma_bf(af, bf, acc[rt]);
        }
    }
    int n  = n0 + (lane & 15);
    int mb = (lane >> 4) << 3;
    float bv = bias[n];
#pragma unroll
    for (int rt = 0; rt < ROWT; ++rt)
#pragma unroll
        for (int r = 0; r < 8; ++r) {
            float v = acc[rt][r] + bv;
            v = fminf(fmaxf(v, -4.f), 3.f);
            atomicAdd(&logsum[rt * 16 + mb + r], v);
        }
}

// ---------------- kernels ----------------

// f32 [Ksrc x Nsrc] row-major -> bf16 transposed [Ndst][Kpad], optional col slice
__global__ void prep_wT(const float* __restrict__ src, unsigned short* __restrict__ dst,
                        int Ksrc, int Kpad, int Ndst, int colOff, int Nsrc) {
    int i = blockIdx.x * blockDim.x + threadIdx.x;
    if (i >= Ndst * Kpad) return;
    int n = i / Kpad, k = i % Kpad;
    float v = 0.f;
    int c = n + colOff;
    if (k < Ksrc && c < Nsrc) v = src[(size_t)k * Nsrc + c];
    dst[i] = f2bf(v);
}

__global__ __launch_bounds__(256)
void encoder_kernel(const float* __restrict__ s0,
                    const float* __restrict__ enc_b1, const float* __restrict__ enc_b2,
                    const float* __restrict__ zmu_b,  const float* __restrict__ zls_b,
                    const unsigned short* __restrict__ W,
                    float* __restrict__ muv, float* __restrict__ stdv) {
    __shared__ __align__(16) unsigned short sXA[TM * 64];
    __shared__ __align__(16) unsigned short sA0[TM * 256];
    __shared__ __align__(16) unsigned short sA1[TM * 256];
    __shared__ float sMu[TM * 16];
    __shared__ float sLs[TM * 16];

    int tid = threadIdx.x, lane = tid & 31, wave = tid >> 5;
    int row0 = blockIdx.x * TM;

    for (int j = tid; j < TM * 64; j += 256) {
        int m = j >> 6, c = j & 63;
        sXA[m * 64 + c] = f2bf(s0[(size_t)(row0 + m) * OBS_ + c]);
    }
    __syncthreads();
    layer_relu(sXA, W + OFF_ENCW1, enc_b1, sA0, 64, 2, 256, 256, lane, wave);
    __syncthreads();
    layer_relu(sA0, W + OFF_ENCW2, enc_b2, sA1, 256, 8, 256, 256, lane, wave);
    __syncthreads();
    if (wave < 4) head_f32(sA1, W + OFF_ZMU, zmu_b, 16, sMu, 256, 8, wave, lane);
    else          head_f32(sA1, W + OFF_ZLS, zls_b, 16, sLs, 256, 8, wave - 4, lane);
    __syncthreads();
    for (int j = tid; j < TM * 16; j += 256) {
        int m = j >> 4, d = j & 15;
        int r = row0 + m;
        muv[r * ZD_ + d] = sMu[m * 16 + d];
        float ls = fminf(fmaxf(sLs[m * 16 + d], -4.f), 2.f);
        stdv[r * ZD_ + d] = expf(ls);
    }
}

__global__ __launch_bounds__(256)
void score_kernel(const float* __restrict__ s0, const float* __restrict__ eps,
                  const float* __restrict__ mu_b1, const float* __restrict__ mu_b2,
                  const float* __restrict__ mu_b3,
                  const float* __restrict__ sig_b1, const float* __restrict__ sig_b2,
                  const float* __restrict__ sig_b3,
                  const float* __restrict__ risk_b1, const float* __restrict__ risk_b2,
                  const float* __restrict__ risk_b3,
                  const unsigned short* __restrict__ W,
                  const float* __restrict__ muv, const float* __restrict__ stdv,
                  float* __restrict__ score, int iter) {
    __shared__ __align__(16) unsigned short sXA[TM * 96];   // [s0 | z | pad]
    __shared__ __align__(16) unsigned short sZA[TM * 32];   // [z | pad]
    __shared__ __align__(16) unsigned short sA0[TM * 256];
    __shared__ __align__(16) unsigned short sA1[TM * 256];
    __shared__ float sLog[TM];
    __shared__ float sMu3[TM * 16];
    __shared__ float sRk [TM * 16];

    int tid = threadIdx.x, lane = tid & 31, wave = tid >> 5;
    int row0 = blockIdx.x * TM;
    int b = row0 >> 9;                       // 64 | 512, so one b per WG

    for (int j = tid; j < TM * 64; j += 256) {
        int m = j >> 6, c = j & 63;
        sXA[m * 96 + c] = f2bf(s0[(size_t)b * OBS_ + c]);
    }
    for (int j = tid; j < TM * 16; j += 256) {
        int m = j >> 4, d = j & 15;
        int s = (row0 + m) & (S_ - 1);
        float e = eps[((((size_t)iter * B_ + b) * S_) + s) * ZD_ + d];
        unsigned short z = f2bf(muv[b * ZD_ + d] + stdv[b * ZD_ + d] * e);
        sXA[m * 96 + 64 + d] = z;
        sXA[m * 96 + 80 + d] = 0;
        sZA[m * 32 + d]      = z;
        sZA[m * 32 + 16 + d] = 0;
    }
    if (tid < TM) sLog[tid] = 0.f;
    __syncthreads();

    // mu branch
    layer_relu(sXA, W + OFF_MUW1, mu_b1, sA0, 96, 3, 256, 256, lane, wave);
    __syncthreads();
    layer_relu(sA0, W + OFF_MUW2, mu_b2, sA1, 256, 8, 256, 256, lane, wave);
    __syncthreads();
    if (wave < 4) head_f32(sA1, W + OFF_MUW3, mu_b3 + 112, 16, sMu3, 256, 8, wave, lane);
    __syncthreads();

    // sig branch
    layer_relu(sZA, W + OFF_SIGW1, sig_b1, sA0, 32, 1, 256, 256, lane, wave);
    __syncthreads();
    layer_relu(sA0, W + OFF_SIGW2, sig_b2, sA1, 256, 8, 256, 256, lane, wave);
    __syncthreads();
    layer_sig3(sA1, W + OFF_SIGW3, sig_b3, sLog, lane, wave);
    __syncthreads();

    // risk branch
    layer_relu(sXA, W + OFF_RKW1, risk_b1, sA0, 96, 3, 256, 256, lane, wave);
    __syncthreads();
    layer_relu(sA0, W + OFF_RKW2, risk_b2, sA1, 256, 8, 256, 256, lane, wave);
    __syncthreads();
    if (wave >= 4) head_f32(sA1, W + OFF_RKW3, risk_b3, 1, sRk, 256, 8, wave - 4, lane);
    __syncthreads();

    if (tid < TM) {
        int m = tid;
        int s = (row0 + m) & (S_ - 1);
        float g0 = s0[(size_t)b * OBS_ + 2], g1 = s0[(size_t)b * OBS_ + 3];
        float d0 = sMu3[m * 16 + 14] - g0;   // mu_out col 126
        float d1 = sMu3[m * 16 + 15] - g1;   // mu_out col 127
        float intent = d0 * d0 + d1 * d1;
        float agency = -sLog[m] * (1.f / 128.f);
        float rr = sRk[m * 16 + 0];
        float risk = 1.f / (1.f + expf(-rr));
        score[(size_t)b * S_ + s] = -intent + 0.5f * agency - risk;
    }
}

__global__ __launch_bounds__(256)
void cem_update_kernel(const float* __restrict__ eps, const float* __restrict__ score,
                       float* __restrict__ muv, float* __restrict__ stdv,
                       float* __restrict__ bestz, float* __restrict__ bests,
                       float* __restrict__ out, int iter) {
    __shared__ float sc[S_];
    __shared__ float rv[256];
    __shared__ int   ri[256];
    __shared__ int   elite[ELITES_];
    __shared__ float sTop;

    int b = blockIdx.x, tid = threadIdx.x;
    sc[tid]       = score[(size_t)b * S_ + tid];
    sc[tid + 256] = score[(size_t)b * S_ + tid + 256];

    float muOld = 0.f, stdOld = 0.f;
    if (tid < ZD_) { muOld = muv[b * ZD_ + tid]; stdOld = stdv[b * ZD_ + tid]; }
    __syncthreads();

    for (int e = 0; e < ELITES_; ++e) {
        float v1 = sc[tid], v2 = sc[tid + 256];
        float v; int idx;
        if (v2 > v1) { v = v2; idx = tid + 256; } else { v = v1; idx = tid; }
        rv[tid] = v; ri[tid] = idx;
        __syncthreads();
        for (int off = 128; off > 0; off >>= 1) {
            if (tid < off) {
                float vo = rv[tid + off]; int io = ri[tid + off];
                if (vo > rv[tid] || (vo == rv[tid] && io < ri[tid])) { rv[tid] = vo; ri[tid] = io; }
            }
            __syncthreads();
        }
        if (tid == 0) {
            elite[e] = ri[0];
            if (e == 0) sTop = rv[0];
            sc[ri[0]] = -INFINITY;
        }
        __syncthreads();
    }

    if (tid < ZD_) {
        int d = tid;
        float sum = 0.f, sumsq = 0.f, ztop = 0.f;
        for (int e = 0; e < ELITES_; ++e) {
            int s = elite[e];
            float ev = eps[((((size_t)iter * B_ + b) * S_) + s) * ZD_ + d];
            float z = muOld + stdOld * ev;
            if (e == 0) ztop = z;
            sum += z; sumsq += z * z;
        }
        float mean = sum * (1.f / 64.f);
        float var  = fmaxf(sumsq - 64.f * mean * mean, 0.f) * (1.f / 63.f);
        float sstd = sqrtf(var);
        muv[b * ZD_ + d]  = 0.25f * muOld + 0.75f * mean;
        stdv[b * ZD_ + d] = fmaxf(0.25f * stdOld + 0.75f * sstd, 0.2f);

        float top = sTop;
        float prevBest = (iter == 0) ? -INFINITY : bests[b];
        bool better = top > prevBest;
        float bz = (iter == 0) ? ztop : (better ? ztop : bestz[b * ZD_ + d]);
        float bs = better ? top : prevBest;
        bestz[b * ZD_ + d] = bz;
        if (d == 0) bests[b] = bs;
        if (iter == ITERS_ - 1) {
            out[b * 17 + d] = bz;
            if (d == 0) out[b * 17 + 16] = bs;
        }
    }
}

// ---------------- launcher ----------------
extern "C" void kernel_launch(void* const* d_in, const int* in_sizes, int n_in,
                              void* d_out, int out_size, void* d_ws, size_t ws_size,
                              hipStream_t stream) {
    (void)in_sizes; (void)n_in; (void)out_size; (void)ws_size;

    const float* s0      = (const float*)d_in[0];
    const float* eps     = (const float*)d_in[1];
    const float* enc_w1  = (const float*)d_in[2];
    const float* enc_b1  = (const float*)d_in[3];
    const float* enc_w2  = (const float*)d_in[4];
    const float* enc_b2  = (const float*)d_in[5];
    const float* zmu_w   = (const float*)d_in[6];
    const float* zmu_b   = (const float*)d_in[7];
    const float* zls_w   = (const float*)d_in[8];
    const float* zls_b   = (const float*)d_in[9];
    const float* mu_w1   = (const float*)d_in[10];
    const float* mu_b1   = (const float*)d_in[11];
    const float* mu_w2   = (const float*)d_in[12];
    const float* mu_b2   = (const float*)d_in[13];
    const float* mu_w3   = (const float*)d_in[14];
    const float* mu_b3   = (const float*)d_in[15];
    const float* sig_w1  = (const float*)d_in[16];
    const float* sig_b1  = (const float*)d_in[17];
    const float* sig_w2  = (const float*)d_in[18];
    const float* sig_b2  = (const float*)d_in[19];
    const float* sig_w3  = (const float*)d_in[20];
    const float* sig_b3  = (const float*)d_in[21];
    const float* risk_w1 = (const float*)d_in[22];
    const float* risk_b1 = (const float*)d_in[23];
    const float* risk_w2 = (const float*)d_in[24];
    const float* risk_b2 = (const float*)d_in[25];
    const float* risk_w3 = (const float*)d_in[26];
    const float* risk_b3 = (const float*)d_in[27];

    unsigned short* W = (unsigned short*)d_ws;
    float* F = (float*)((char*)d_ws + (size_t)USH_TOTAL * sizeof(unsigned short));
    float* muv   = F + FOFF_MU;
    float* stdv  = F + FOFF_STD;
    float* bestz = F + FOFF_BESTZ;
    float* bests = F + FOFF_BESTS;
    float* scorep = F + FOFF_SCORE;

    struct PrepArg { const float* src; size_t off; int Ksrc, Kpad, Ndst, colOff, Nsrc; };
    const PrepArg preps[] = {
        { enc_w1,  OFF_ENCW1,  64,  64, 256,   0, 256 },
        { enc_w2,  OFF_ENCW2, 256, 256, 256,   0, 256 },
        { zmu_w,   OFF_ZMU,   256, 256,  16,   0,  16 },
        { zls_w,   OFF_ZLS,   256, 256,  16,   0,  16 },
        { mu_w1,   OFF_MUW1,   80,  96, 256,   0, 256 },
        { mu_w2,   OFF_MUW2,  256, 256, 256,   0, 256 },
        { mu_w3,   OFF_MUW3,  256, 256,  16, 112, 128 },
        { sig_w1,  OFF_SIGW1,  16,  32, 256,   0, 256 },
        { sig_w2,  OFF_SIGW2, 256, 256, 256,   0, 256 },
        { sig_w3,  OFF_SIGW3, 256, 256, 128,   0, 128 },
        { risk_w1, OFF_RKW1,   80,  96, 256,   0, 256 },
        { risk_w2, OFF_RKW2,  256, 256, 256,   0, 256 },
        { risk_w3, OFF_RKW3,  256, 256,  16,   0,   1 },
    };
    for (int p = 0; p < 13; ++p) {
        int total = preps[p].Ndst * preps[p].Kpad;
        prep_wT<<<(total + 255) / 256, 256, 0, stream>>>(
            preps[p].src, W + preps[p].off,
            preps[p].Ksrc, preps[p].Kpad, preps[p].Ndst, preps[p].colOff, preps[p].Nsrc);
    }

    encoder_kernel<<<B_ / TM, 256, 0, stream>>>(s0, enc_b1, enc_b2, zmu_b, zls_b,
                                                W, muv, stdv);

    for (int i = 0; i < ITERS_; ++i) {
        score_kernel<<<(B_ * S_) / TM, 256, 0, stream>>>(
            s0, eps, mu_b1, mu_b2, mu_b3, sig_b1, sig_b2, sig_b3,
            risk_b1, risk_b2, risk_b3, W, muv, stdv, scorep, i);
        cem_update_kernel<<<B_, 256, 0, stream>>>(
            eps, scorep, muv, stdv, bestz, bests, (float*)d_out, i);
    }
}
